// GCN_47828755808670
// MI455X (gfx1250) — matrix operand order repaired and checked by reading, not compiled, not run on hardware
//
#include <hip/hip_runtime.h>
#include <hip/hip_fp16.h>
#include <stdint.h>

typedef __attribute__((ext_vector_type(16))) _Float16 v16h;
typedef __attribute__((ext_vector_type(8)))  _Float16 v8h;
typedef __attribute__((ext_vector_type(4)))  _Float16 v4h;
typedef __attribute__((ext_vector_type(8)))  float    v8f;
typedef __attribute__((ext_vector_type(4)))  int      v4i;

#define NN 100000
#define EE 1600000
#define HH 128
#define CC 10
#define GG 128

#if defined(__gfx1250__) && __has_builtin(__builtin_amdgcn_global_load_async_to_lds_b128)
#define HAVE_ASYNC_LDS 1
#endif

// ---------------------------------------------------------------------------
// Weight prep: convert f32 row-major W[k][n] -> f16 col-major Wt[l][n][k]
// so each WMMA lane can load its 16 contiguous halves (32B aligned).
// ---------------------------------------------------------------------------
__global__ void prep_weights(const float* __restrict__ w_init,
                             const float* __restrict__ w_conv,
                             _Float16* __restrict__ wt) {
    int bid = blockIdx.x;          // l*HH + n
    int l = bid >> 7;
    int n = bid & 127;
    int k = threadIdx.x;
    const float* W = (l == 0) ? w_init : (w_conv + (size_t)(l - 1) * HH * HH);
    wt[(size_t)l * HH * HH + (size_t)n * HH + k] = (_Float16)W[(size_t)k * HH + n];
}

__global__ void cast_to_f16(const float* __restrict__ src,
                            _Float16* __restrict__ dst, int n4) {
    int i = blockIdx.x * blockDim.x + threadIdx.x;
    if (i >= n4) return;
    float4 v = ((const float4*)src)[i];
    v4h h = { (_Float16)v.x, (_Float16)v.y, (_Float16)v.z, (_Float16)v.w };
    ((v4h*)dst)[i] = h;
}

// ---------------------------------------------------------------------------
// Degree / normalization precompute
// ---------------------------------------------------------------------------
__global__ void deg_init(float* __restrict__ deg) {
    int i = blockIdx.x * blockDim.x + threadIdx.x;
    if (i < NN) deg[i] = 1.0f;            // self-loop
}
__global__ void deg_edges(const long long* __restrict__ ei, float* __restrict__ deg) {
    int e = blockIdx.x * blockDim.x + threadIdx.x;
    if (e < EE) atomicAdd(&deg[(int)ei[EE + e]], 1.0f);
}
__global__ void dinv_fin(float* __restrict__ deg) {
    int i = blockIdx.x * blockDim.x + threadIdx.x;
    if (i < NN) { float d = deg[i]; deg[i] = (d > 0.f) ? rsqrtf(d) : 0.f; }
}
__global__ void norm_edges(const long long* __restrict__ ei,
                           const float* __restrict__ dinv,
                           float* __restrict__ norm) {
    int e = blockIdx.x * blockDim.x + threadIdx.x;
    if (e < EE) norm[e] = dinv[(int)ei[e]] * dinv[(int)ei[EE + e]];
}

// ---------------------------------------------------------------------------
// WMMA GEMM with LDS-staged B: C[N,128] = A16[N,128] @ W.
// Each 256-thread block stages the whole 128x128 f16 weight tile (32 KB) in
// LDS (async direct-to-LDS when available); its 8 waves each produce one
// 16-row strip. Inner loop batches 8 B fragments per K-step so 8 WMMAs issue
// back-to-back behind a single LDS wait.
// ---------------------------------------------------------------------------
__global__ __launch_bounds__(256) void gemm128_wmma(
    const _Float16* __restrict__ A, const _Float16* __restrict__ Bt,
    float* __restrict__ Cmat, int nrows) {
    __shared__ alignas(16) _Float16 Bs[HH * HH];     // 32 KB

#if defined(HAVE_ASYNC_LDS)
    // Async global->LDS DMA copies (ASYNCcnt path), 2048 x 16B.
    for (int t = threadIdx.x; t < HH * HH / 8; t += 256) {
        __builtin_amdgcn_global_load_async_to_lds_b128(
            (v4i*)(Bt + (size_t)t * 8),
            (v4i*)(Bs + (size_t)t * 8),
            0, 0);
    }
#if __has_builtin(__builtin_amdgcn_s_wait_asynccnt)
    __builtin_amdgcn_s_wait_asynccnt(0);
#else
    asm volatile("s_wait_asynccnt 0x0" ::: "memory");
#endif
#else
    for (int t = threadIdx.x; t < HH * HH / 8; t += 256) {
        ((v8h*)Bs)[t] = ((const v8h*)Bt)[t];
    }
#endif
    __syncthreads();

    int wave = (blockIdx.x * blockDim.x + threadIdx.x) >> 5;
    int lane = threadIdx.x & 31;
    int r0 = wave * 16;
    if (r0 >= nrows) return;
    int lm = lane & 15;        // row within tile (A) / col within tile (B,D)
    int lg = lane >> 4;        // lane group

    // A fragments: 16x32 layout, lane holds K [g*16..g*16+15]
    // of row M=lm -> two contiguous 16B loads per step.
    const _Float16* arow = A + (size_t)(r0 + lm) * HH;
    v16h afrag[4];
#pragma unroll
    for (int kk = 0; kk < 4; ++kk) {
        int kb = kk * 32 + lg * 16;
        v8h lo = *(const v8h*)(arow + kb);
        v8h hi = *(const v8h*)(arow + kb + 8);
        v16h a;
#pragma unroll
        for (int i = 0; i < 8; ++i) { a[i] = lo[i]; a[i + 8] = hi[i]; }
        afrag[kk] = a;
    }

    v8f acc[8] = {};
#pragma unroll
    for (int kk = 0; kk < 4; ++kk) {
        int kb = kk * 32 + lg * 16;
        v16h bfrag[8];
#pragma unroll
        for (int ct = 0; ct < 8; ++ct) {
            // B 32x16 layout: lane holds col=lm, K = kb..kb+15 contiguous
            bfrag[ct] = *(const v16h*)(Bs + (size_t)(ct * 16 + lm) * HH + kb);
        }
#pragma unroll
        for (int ct = 0; ct < 8; ++ct) {
            acc[ct] = __builtin_amdgcn_wmma_f32_16x16x32_f16(
                false, afrag[kk], false, bfrag[ct], (short)0, acc[ct], false, false);
        }
    }

    // D layout: lane holds col=ct*16+lm, VGPR v = row r0 + v + 8*lg
#pragma unroll
    for (int ct = 0; ct < 8; ++ct) {
        float* crow = Cmat + (size_t)(r0 + lg * 8) * HH + (size_t)ct * 16 + lm;
#pragma unroll
        for (int v = 0; v < 8; ++v) crow[(size_t)v * HH] = acc[ct][v];
    }
}

// ---------------------------------------------------------------------------
// Aggregation: agg = b + dinv^2 * hw (self loop), then edge scatter-add.
// ---------------------------------------------------------------------------
__global__ void agg_init(const float* __restrict__ hw,
                         const float* __restrict__ dinv,
                         const float* __restrict__ bias,
                         float* __restrict__ agg) {
    int idx = blockIdx.x * blockDim.x + threadIdx.x;   // over NH/4
    if (idx >= NN * HH / 4) return;
    int i = idx >> 5;                                  // node
    int f4 = idx & 31;                                 // feature quad
    float dv = dinv[i];
    float d2 = dv * dv;
    float4 h = ((const float4*)hw)[idx];
    float4 b = ((const float4*)bias)[f4];
    float4 r = { b.x + d2 * h.x, b.y + d2 * h.y, b.z + d2 * h.z, b.w + d2 * h.w };
    ((float4*)agg)[idx] = r;
}

// One wave per edge: edge metadata via SGPRs (s_load), each lane owns a
// float4 of features -> 1 b128 load + 4 f32 atomics per lane.
__global__ __launch_bounds__(256) void scatter_edges(
    const float* __restrict__ hw, float* __restrict__ agg,
    const long long* __restrict__ ei, const float* __restrict__ norm) {
    int e = __builtin_amdgcn_readfirstlane(
        (int)((blockIdx.x * blockDim.x + threadIdx.x) >> 5));
    int lane = threadIdx.x & 31;
    if (e >= EE) return;
    int s = (int)ei[e];            // wave-uniform -> scalar load
    int d = (int)ei[EE + e];
    float nm = norm[e];
    float4 v = *(const float4*)(hw + (size_t)s * HH + lane * 4);
    float* dst = agg + (size_t)d * HH + lane * 4;
    atomicAdd(dst + 0, nm * v.x);
    atomicAdd(dst + 1, nm * v.y);
    atomicAdd(dst + 2, nm * v.z);
    atomicAdd(dst + 3, nm * v.w);
}

__global__ void relu_cast(float* __restrict__ agg, _Float16* __restrict__ h16) {
    int idx = blockIdx.x * blockDim.x + threadIdx.x;   // over NH/4
    if (idx >= NN * HH / 4) return;
    float4 v = ((const float4*)agg)[idx];
    v.x = fmaxf(v.x, 0.f); v.y = fmaxf(v.y, 0.f);
    v.z = fmaxf(v.z, 0.f); v.w = fmaxf(v.w, 0.f);
    ((float4*)agg)[idx] = v;
    v4h h = { (_Float16)v.x, (_Float16)v.y, (_Float16)v.z, (_Float16)v.w };
    ((v4h*)h16)[idx] = h;
}

// ---------------------------------------------------------------------------
// Global mean pool: one wave per node, batch id via scalar load.
// ---------------------------------------------------------------------------
__global__ void pool_zero(float* __restrict__ pooled, float* __restrict__ cnt) {
    int i = blockIdx.x * blockDim.x + threadIdx.x;
    if (i < GG * HH) pooled[i] = 0.0f;
    if (i < GG) cnt[i] = 0.0f;
}
__global__ __launch_bounds__(256) void pool_sum(
    const float* __restrict__ h, const long long* __restrict__ batch,
    float* __restrict__ pooled) {
    int i = __builtin_amdgcn_readfirstlane(
        (int)((blockIdx.x * blockDim.x + threadIdx.x) >> 5));
    int lane = threadIdx.x & 31;
    if (i >= NN) return;
    int g = (int)batch[i];         // wave-uniform -> scalar load
    float4 v = *(const float4*)(h + (size_t)i * HH + lane * 4);
    float* dst = pooled + (size_t)g * HH + lane * 4;
    atomicAdd(dst + 0, v.x);
    atomicAdd(dst + 1, v.y);
    atomicAdd(dst + 2, v.z);
    atomicAdd(dst + 3, v.w);
}
__global__ void pool_cnt(const long long* __restrict__ batch, float* __restrict__ cnt) {
    int i = blockIdx.x * blockDim.x + threadIdx.x;
    if (i < NN) atomicAdd(&cnt[(int)batch[i]], 1.0f);
}

// ---------------------------------------------------------------------------
// Head: pooled/cnt -> relu(pooled@w1+b1) -> @w2+b2 -> log_softmax
// ---------------------------------------------------------------------------
__global__ __launch_bounds__(128) void head_kernel(
    const float* __restrict__ pooled, const float* __restrict__ cnt,
    const float* __restrict__ w1, const float* __restrict__ b1,
    const float* __restrict__ w2, const float* __restrict__ b2,
    float* __restrict__ out) {
    __shared__ float p[HH];
    __shared__ float hid[HH];
    __shared__ float lg[CC];
    int g = blockIdx.x, t = threadIdx.x;
    float c = fmaxf(cnt[g], 1.0f);
    p[t] = pooled[(size_t)g * HH + t] / c;
    __syncthreads();
    float acc = b1[t];
    for (int k = 0; k < HH; ++k) acc += p[k] * w1[(size_t)k * HH + t];
    hid[t] = fmaxf(acc, 0.0f);
    __syncthreads();
    if (t < CC) {
        float a = b2[t];
        for (int k = 0; k < HH; ++k) a += hid[k] * w2[(size_t)k * CC + t];
        lg[t] = a;
    }
    __syncthreads();
    if (t == 0) {
        float m = lg[0];
        for (int i = 1; i < CC; ++i) m = fmaxf(m, lg[i]);
        float s = 0.0f;
        for (int i = 0; i < CC; ++i) s += expf(lg[i] - m);
        float lse = m + logf(s);
        for (int i = 0; i < CC; ++i) out[(size_t)g * CC + i] = lg[i] - lse;
    }
}

// ---------------------------------------------------------------------------
// Launch
// ---------------------------------------------------------------------------
static inline size_t align256(size_t x) { return (x + 255) & ~(size_t)255; }

extern "C" void kernel_launch(void* const* d_in, const int* in_sizes, int n_in,
                              void* d_out, int out_size, void* d_ws, size_t ws_size,
                              hipStream_t stream) {
    const float*     x      = (const float*)d_in[0];
    const long long* ei     = (const long long*)d_in[1];
    const long long* batch  = (const long long*)d_in[2];
    const float*     w_init = (const float*)d_in[3];
    const float*     b_init = (const float*)d_in[4];
    const float*     w_conv = (const float*)d_in[5];
    const float*     b_conv = (const float*)d_in[6];
    const float*     w_h1   = (const float*)d_in[7];
    const float*     b_h1   = (const float*)d_in[8];
    const float*     w_h2   = (const float*)d_in[9];
    const float*     b_h2   = (const float*)d_in[10];
    float* out = (float*)d_out;

    char* ws = (char*)d_ws;
    size_t off = 0;
    float*    dinv   = (float*)(ws + off); off = align256(off + (size_t)NN * 4);
    float*    norm   = (float*)(ws + off); off = align256(off + (size_t)EE * 4);
    _Float16* h16    = (_Float16*)(ws + off); off = align256(off + (size_t)NN * HH * 2);
    float*    hw     = (float*)(ws + off); off = align256(off + (size_t)NN * HH * 4);
    float*    agg    = (float*)(ws + off); off = align256(off + (size_t)NN * HH * 4);
    _Float16* wt16   = (_Float16*)(ws + off); off = align256(off + (size_t)3 * HH * HH * 2);
    float*    pooled = (float*)(ws + off); off = align256(off + (size_t)GG * HH * 4);
    float*    cnt    = (float*)(ws + off); off = align256(off + (size_t)GG * 4);
    (void)ws_size; (void)n_in; (void)in_sizes; (void)out_size;

    const int T = 256;
    const int NH = NN * HH;
    const int NH4 = NH / 4;

    // Precompute: weights, f16 input, degree norms
    prep_weights<<<3 * HH, HH, 0, stream>>>(w_init, w_conv, wt16);
    cast_to_f16<<<(NH4 + T - 1) / T, T, 0, stream>>>(x, h16, NH4);
    deg_init<<<(NN + T - 1) / T, T, 0, stream>>>(dinv);
    deg_edges<<<(EE + T - 1) / T, T, 0, stream>>>(ei, dinv);
    dinv_fin<<<(NN + T - 1) / T, T, 0, stream>>>(dinv);
    norm_edges<<<(EE + T - 1) / T, T, 0, stream>>>(ei, dinv, norm);

    // 3 GCN layers: wmma matmul -> self-loop+bias init -> edge scatter -> relu
    const int waves = NN / 16;                       // 6250, exact
    const int gemm_blocks = (waves * 32 + T - 1) / T;
    for (int l = 0; l < 3; ++l) {
        const float* bias = (l == 0) ? b_init : (b_conv + (size_t)(l - 1) * HH);
        gemm128_wmma<<<gemm_blocks, T, 0, stream>>>(h16, wt16 + (size_t)l * HH * HH,
                                                    hw, NN);
        agg_init<<<(NH4 + T - 1) / T, T, 0, stream>>>(hw, dinv, bias, agg);
        scatter_edges<<<EE / 8, T, 0, stream>>>(hw, agg, ei, norm);
        relu_cast<<<(NH4 + T - 1) / T, T, 0, stream>>>(agg, h16);
    }

    // Mean pool + head
    pool_zero<<<(GG * HH + T - 1) / T, T, 0, stream>>>(pooled, cnt);
    pool_sum<<<(NN * 32 + T - 1) / T, T, 0, stream>>>(agg, batch, pooled);
    pool_cnt<<<(NN + T - 1) / T, T, 0, stream>>>(batch, cnt);
    head_kernel<<<GG, HH, 0, stream>>>(pooled, cnt, w_h1, b_h1, w_h2, b_h2, out);
}
